// GPRGNN_57097295233450
// MI455X (gfx1250) — matrix operand order repaired and checked by reading, not compiled, not run on hardware
//
#include <hip/hip_runtime.h>

typedef __attribute__((ext_vector_type(16))) __bf16 v16bf;
typedef __attribute__((ext_vector_type(8)))  __bf16 v8bf;
typedef __attribute__((ext_vector_type(8)))  float  v8f;

#define N_FIN 512
#define N_H   256
#define N_C   64
#define N_K   10

static __device__ __forceinline__ __bf16 f2bf(float f) { return (__bf16)f; }

// ---------------------------------------------------------------------------
// GEMM1: h1 = relu(x @ w1^T + b1), x:[N,512] f32, w1t:[512,256] bf16 (pre-
// transposed), out h1b:[N,256] bf16.  128x64 block tile, 8 waves of 2x2
// 16x16x32 bf16 WMMA, K staged through LDS with f32->bf16 convert.
// ---------------------------------------------------------------------------
__global__ __launch_bounds__(256) void k_gemm1(
    const float* __restrict__ x, const __bf16* __restrict__ w1t,
    const float* __restrict__ b1, __bf16* __restrict__ h1b, int N)
{
  constexpr int TM = 128, TK = 32, LDA = 40;   // pad 32->40 (80B row stride)
  __shared__ __align__(128) __bf16 As[TM * LDA];

  const int mb   = blockIdx.x * TM;
  const int nb   = blockIdx.y * 64;
  const int tid  = threadIdx.x;
  const int wave = tid >> 5;
  const int lane = tid & 31;
  const int wr   = wave >> 1;            // 0..3  (32-row strip)
  const int wc   = wave & 1;             // 0..1  (32-col strip)
  const int am    = lane & 15;           // A: M index
  const int khalf = (lane >> 4) << 3;    // A: 0 or 8 (K-half select)

  const int lrow = tid >> 1;             // LDS stage-in: row 0..127
  const int lcol = (tid & 1) << 4;       // cols 0..15 or 16..31

  v8f acc[2][2] = {};

  for (int kb = 0; kb < N_FIN; kb += TK) {
    // ---- stage x tile (f32 -> bf16) into LDS --------------------------------
    {
      const int grow = mb + lrow;
      __bf16* dst = &As[lrow * LDA + lcol];
      v8bf o0, o1;
      if (grow < N) {
        const float4* xp = (const float4*)(x + (size_t)grow * N_FIN + kb + lcol);
        float4 f0 = xp[0], f1 = xp[1], f2 = xp[2], f3 = xp[3];
        o0[0]=f2bf(f0.x); o0[1]=f2bf(f0.y); o0[2]=f2bf(f0.z); o0[3]=f2bf(f0.w);
        o0[4]=f2bf(f1.x); o0[5]=f2bf(f1.y); o0[6]=f2bf(f1.z); o0[7]=f2bf(f1.w);
        o1[0]=f2bf(f2.x); o1[1]=f2bf(f2.y); o1[2]=f2bf(f2.z); o1[3]=f2bf(f2.w);
        o1[4]=f2bf(f3.x); o1[5]=f2bf(f3.y); o1[6]=f2bf(f3.z); o1[7]=f2bf(f3.w);
      } else {
        #pragma unroll
        for (int i = 0; i < 8; ++i) { o0[i] = f2bf(0.f); o1[i] = f2bf(0.f); }
      }
      *(v8bf*)dst       = o0;
      *(v8bf*)(dst + 8) = o1;
    }
    __syncthreads();

    // ---- B fragments: lane%16 = N col, 16 K elements in table order ---------
    v16bf bfrag[2];
    #pragma unroll
    for (int c = 0; c < 2; ++c) {
      v16bf t;
      #pragma unroll
      for (int i = 0; i < 16; ++i)
        t[i] = w1t[(size_t)(kb + khalf + (i & 7) + ((i >> 3) << 4)) * N_H + nb + wc * 32 + c * 16 + am];
      bfrag[c] = t;
    }

    // ---- A fragments from LDS ----------------------------------------------
    v16bf afrag[2];
    #pragma unroll
    for (int r = 0; r < 2; ++r) {
      const __bf16* ap = &As[(wr * 32 + r * 16 + am) * LDA];
      v8bf lo = *(const v8bf*)(ap + khalf);
      v8bf hi = *(const v8bf*)(ap + 16 + khalf);
      v16bf t;
      #pragma unroll
      for (int i = 0; i < 8; ++i) { t[i] = lo[i]; t[i + 8] = hi[i]; }
      afrag[r] = t;
    }

    #pragma unroll
    for (int r = 0; r < 2; ++r)
      #pragma unroll
      for (int c = 0; c < 2; ++c)
        acc[r][c] = __builtin_amdgcn_wmma_f32_16x16x32_bf16(
            false, afrag[r], false, bfrag[c], (short)0, acc[r][c], false, false);
    __syncthreads();
  }

  // ---- epilogue: relu(acc + b1) -> bf16 ------------------------------------
  #pragma unroll
  for (int r = 0; r < 2; ++r) {
    #pragma unroll
    for (int c = 0; c < 2; ++c) {
      const int col = nb + wc * 32 + c * 16 + am;
      const float bias = b1[col];
      #pragma unroll
      for (int v = 0; v < 8; ++v) {
        const int row = mb + wr * 32 + r * 16 + v + ((lane >> 4) << 3);
        if (row < N)
          h1b[(size_t)row * N_H + col] = f2bf(fmaxf(acc[r][c][v] + bias, 0.f));
      }
    }
  }
}

// ---------------------------------------------------------------------------
// GEMM2: h = h1 @ w2^T + b2 -> hcur [N,64] f32; also hidden = temp[0]*h.
// A already bf16 in global, B is tiny (32KB, L2-resident): direct loads.
// ---------------------------------------------------------------------------
__global__ __launch_bounds__(256) void k_gemm2(
    const __bf16* __restrict__ h1b, const __bf16* __restrict__ w2t,
    const float* __restrict__ b2, const float* __restrict__ temp,
    float* __restrict__ hcur, float* __restrict__ hidden, int N)
{
  const int mb   = blockIdx.x * 128;
  const int tid  = threadIdx.x;
  const int wave = tid >> 5, lane = tid & 31;
  const int wr = wave >> 1, wc = wave & 1;
  const int am = lane & 15, khalf = (lane >> 4) << 3;
  const float t0 = temp[0];

  v8f acc[2][2] = {};

  for (int kb = 0; kb < N_H; kb += 32) {
    v16bf bfrag[2];
    #pragma unroll
    for (int c = 0; c < 2; ++c) {
      v16bf t;
      #pragma unroll
      for (int i = 0; i < 16; ++i)
        t[i] = w2t[(size_t)(kb + khalf + (i & 7) + ((i >> 3) << 4)) * N_C + wc * 32 + c * 16 + am];
      bfrag[c] = t;
    }

    v16bf afrag[2];
    #pragma unroll
    for (int r = 0; r < 2; ++r) {
      const int row = mb + wr * 32 + r * 16 + am;
      const __bf16* ap = h1b + (size_t)(row < N ? row : 0) * N_H + kb;
      v8bf lo = *(const v8bf*)(ap + khalf);
      v8bf hi = *(const v8bf*)(ap + 16 + khalf);
      v16bf t;
      #pragma unroll
      for (int i = 0; i < 8; ++i) { t[i] = lo[i]; t[i + 8] = hi[i]; }
      afrag[r] = t;
    }

    #pragma unroll
    for (int r = 0; r < 2; ++r)
      #pragma unroll
      for (int c = 0; c < 2; ++c)
        acc[r][c] = __builtin_amdgcn_wmma_f32_16x16x32_bf16(
            false, afrag[r], false, bfrag[c], (short)0, acc[r][c], false, false);
  }

  #pragma unroll
  for (int r = 0; r < 2; ++r) {
    #pragma unroll
    for (int c = 0; c < 2; ++c) {
      const int col = wc * 32 + c * 16 + am;
      const float bias = b2[col];
      #pragma unroll
      for (int v = 0; v < 8; ++v) {
        const int row = mb + wr * 32 + r * 16 + v + ((lane >> 4) << 3);
        if (row < N) {
          const float val = acc[r][c][v] + bias;
          const size_t o = (size_t)row * N_C + col;
          hcur[o]   = val;
          hidden[o] = t0 * val;
        }
      }
    }
  }
}

// ---------------------------------------------------------------------------
// Degree / norm precompute
// ---------------------------------------------------------------------------
__global__ void k_fill1(float* __restrict__ deg, int N) {
  int i = blockIdx.x * 256 + threadIdx.x;
  if (i < N) deg[i] = 1.0f;                    // self-loop
}
__global__ void k_count(const long long* __restrict__ dstE, float* __restrict__ deg, int E) {
  int e = blockIdx.x * 256 + threadIdx.x;
  if (e < E) atomicAdd(&deg[(int)dstE[e]], 1.0f);
}
__global__ void k_rsqrt(float* __restrict__ d, int N) {
  int i = blockIdx.x * 256 + threadIdx.x;
  if (i < N) d[i] = rsqrtf(d[i]);              // deg >= 1 always
}
__global__ void k_edges(const long long* __restrict__ srcE, const long long* __restrict__ dstE,
                        const float* __restrict__ dinv, int* __restrict__ s32,
                        int* __restrict__ d32, float* __restrict__ w, int E) {
  int e = blockIdx.x * 256 + threadIdx.x;
  if (e >= E) return;
  int s = (int)srcE[e], d = (int)dstE[e];
  s32[e] = s; d32[e] = d;
  w[e] = dinv[s] * dinv[d];
}

// weight transpose + bf16 convert (one-time, tiny)
__global__ void k_w1t(const float* __restrict__ w1, __bf16* __restrict__ w1t) {
  int idx = blockIdx.x * 256 + threadIdx.x;
  if (idx < N_FIN * N_H) {
    int n = idx & (N_H - 1), k = idx >> 8;
    w1t[idx] = f2bf(w1[(size_t)n * N_FIN + k]);
  }
}
__global__ void k_w2t(const float* __restrict__ w2, __bf16* __restrict__ w2t) {
  int idx = blockIdx.x * 256 + threadIdx.x;
  if (idx < N_H * N_C) {
    int n = idx & (N_C - 1), k = idx >> 6;
    w2t[idx] = f2bf(w2[(size_t)n * N_H + k]);
  }
}

// ---------------------------------------------------------------------------
// Propagation hop: hnext = dinv^2*hcur (self loop) + fused hidden accumulate
// ---------------------------------------------------------------------------
__global__ void k_prep(const float* __restrict__ hcur, float* __restrict__ hnext,
                       float* __restrict__ hidden, const float* __restrict__ dinv,
                       const float* __restrict__ temp, int k, int total) {
  int idx = blockIdx.x * 256 + threadIdx.x;
  if (idx >= total) return;
  const float s  = dinv[idx >> 6];
  const float hv = hcur[idx];
  hnext[idx] = s * s * hv;
  if (k > 0) hidden[idx] += temp[k] * hv;
}

// one wave32 per edge, 2 channels/lane: 256B coalesced gather + f32 atomics
__global__ __launch_bounds__(256) void k_scatter(
    const int* __restrict__ s32, const int* __restrict__ d32,
    const float* __restrict__ w, const float* __restrict__ hcur,
    float* __restrict__ hnext, int E)
{
  int e = blockIdx.x * 8 + (threadIdx.x >> 5);
  if (e >= E) return;
  const int lane = threadIdx.x & 31;
  const int s = s32[e], d = d32[e];
  const float ww = w[e];
  const float2 v = *(const float2*)(hcur + (size_t)s * 64 + lane * 2);
  float* out = hnext + (size_t)d * 64 + lane * 2;
  atomicAdd(out,     ww * v.x);
  atomicAdd(out + 1, ww * v.y);
}

// ---------------------------------------------------------------------------
// final: hidden += temp[K]*h_K, then log_softmax per row (wave per row)
// ---------------------------------------------------------------------------
__global__ __launch_bounds__(256) void k_final(
    const float* __restrict__ hidden, const float* __restrict__ hlast,
    const float* __restrict__ temp, float* __restrict__ out, int N)
{
  int row = blockIdx.x * 8 + (threadIdx.x >> 5);
  if (row >= N) return;
  const int lane = threadIdx.x & 31;
  const float tK = temp[N_K];
  const size_t base = (size_t)row * 64 + lane * 2;
  float v0 = hidden[base]     + tK * hlast[base];
  float v1 = hidden[base + 1] + tK * hlast[base + 1];
  float m = fmaxf(v0, v1);
  #pragma unroll
  for (int off = 16; off > 0; off >>= 1) m = fmaxf(m, __shfl_xor(m, off, 32));
  float e = __expf(v0 - m) + __expf(v1 - m);
  #pragma unroll
  for (int off = 16; off > 0; off >>= 1) e += __shfl_xor(e, off, 32);
  const float lse = m + __logf(e);
  out[base]     = v0 - lse;
  out[base + 1] = v1 - lse;
}

// ---------------------------------------------------------------------------
extern "C" void kernel_launch(void* const* d_in, const int* in_sizes, int n_in,
                              void* d_out, int out_size, void* d_ws, size_t ws_size,
                              hipStream_t stream)
{
  const float* x    = (const float*)d_in[0];
  const float* w1   = (const float*)d_in[1];
  const float* b1   = (const float*)d_in[2];
  const float* w2   = (const float*)d_in[3];
  const float* b2   = (const float*)d_in[4];
  const float* temp = (const float*)d_in[5];
  const long long* ei = (const long long*)d_in[6];   // int64 [2, E]

  const int N = in_sizes[0] / N_FIN;
  const int E = in_sizes[6] / 2;
  const long long* srcE = ei;
  const long long* dstE = ei + E;

  char* ws = (char*)d_ws;
  size_t off = 0;
  auto carve = [&](size_t bytes) -> char* {
    char* p = ws + off;
    off += (bytes + 255) & ~(size_t)255;
    return p;
  };
  float*  dinv   = (float*)carve((size_t)N * 4);
  int*    s32    = (int*)carve((size_t)E * 4);
  int*    d32    = (int*)carve((size_t)E * 4);
  float*  wn     = (float*)carve((size_t)E * 4);
  __bf16* w1t    = (__bf16*)carve((size_t)N_FIN * N_H * 2);
  __bf16* w2t    = (__bf16*)carve((size_t)N_H * N_C * 2);
  __bf16* h1b    = (__bf16*)carve((size_t)N * N_H * 2);
  float*  hA     = (float*)carve((size_t)N * N_C * 4);
  float*  hB     = (float*)carve((size_t)N * N_C * 4);
  float*  hidden = (float*)carve((size_t)N * N_C * 4);

  const dim3 b256(256);
  k_fill1 <<<(N + 255) / 256, b256, 0, stream>>>(dinv, N);
  k_count <<<(E + 255) / 256, b256, 0, stream>>>(dstE, dinv, E);
  k_rsqrt <<<(N + 255) / 256, b256, 0, stream>>>(dinv, N);
  k_edges <<<(E + 255) / 256, b256, 0, stream>>>(srcE, dstE, dinv, s32, d32, wn, E);
  k_w1t   <<<(N_FIN * N_H + 255) / 256, b256, 0, stream>>>(w1, w1t);
  k_w2t   <<<(N_H * N_C + 255) / 256, b256, 0, stream>>>(w2, w2t);

  dim3 g1((N + 127) / 128, N_H / 64);
  k_gemm1 <<<g1, b256, 0, stream>>>(x, w1t, b1, h1b, N);
  k_gemm2 <<<(N + 127) / 128, b256, 0, stream>>>(h1b, w2t, b2, temp, hA, hidden, N);

  float* hcur = hA;
  float* hnext = hB;
  const int total = N * N_C;
  for (int k = 0; k < N_K; ++k) {
    k_prep    <<<(total + 255) / 256, b256, 0, stream>>>(hcur, hnext, hidden, dinv, temp, k, total);
    k_scatter <<<(E + 7) / 8, b256, 0, stream>>>(s32, d32, wn, hcur, hnext, E);
    float* t = hcur; hcur = hnext; hnext = t;
  }
  k_final <<<(N + 7) / 8, b256, 0, stream>>>(hidden, hcur, temp, (float*)d_out, N);
}
